// HybridSurvivalLoss_65747359367312
// MI455X (gfx1250) — compile-verified, matrix operand order checked by reading, not run on hardware
//
#include <hip/hip_runtime.h>
#include <hip/hip_bf16.h>
#include <math.h>

typedef __attribute__((ext_vector_type(16))) __bf16 v16bf;
typedef __attribute__((ext_vector_type(8)))  __bf16 v8bf;
typedef __attribute__((ext_vector_type(8)))  float  v8f;

#define Bn 8192
#define Dn 256
#define NBINS 1000

// workspace layout (float offsets)
#define OFF_XXA   0         // 8192  squared norms z_ct
#define OFF_XXB   8192      // 8192  squared norms z_wsi
#define OFF_ROWA  16384     // 8192  row sums of D_A
#define OFF_ROWB  24576     // 8192  row sums of D_B
#define OFF_ROWAP 32768     // 8*8192 per-jsplit partial row sums A
#define OFF_ROWBP 98304     // 8*8192 per-jsplit partial row sums B
#define OFF_WPART 163840    // 4096*3 per-wave (sAB,sAA,sBB) partials
#define OFF_SCAL  176128    // 32 scalars: [0]=gsumA [1]=gsumB [2]=maxOS [3]=maxRFS
#define OFF_HIST  176160    // 2 endpoints * (S[1024],E[1024],D[1024],R[1024])
#define WS_FLOATS 184352    // ~720 KB base

// staged bf16 copies of the inputs (byte offsets into d_ws)
#define OFF_BF16A_BYTES (1u << 20)                         // 4 MB @ 1 MB
#define OFF_BF16B_BYTES ((1u << 20) + 4u * 1024u * 1024u)  // 4 MB @ 5 MB
#define STAGED_BYTES_NEEDED ((size_t)(1u << 20) + 8u * 1024u * 1024u)

// ---------------- init ----------------
__global__ void zero_kernel(float* __restrict__ ws) {
  for (int i = blockIdx.x * blockDim.x + threadIdx.x + OFF_ROWA; i < WS_FLOATS;
       i += gridDim.x * blockDim.x)
    ws[i] = 0.0f;
}

// ---------------- fp32 -> bf16 staging (8 elems / thread) ----------------
__global__ void cvt_bf16_kernel(const float* __restrict__ X, __bf16* __restrict__ Y) {
  int i = (blockIdx.x * 256 + threadIdx.x) * 8;
  float4 a = *(const float4*)(X + i);
  float4 b = *(const float4*)(X + i + 4);
  v8bf o;
  o[0] = (__bf16)a.x; o[1] = (__bf16)a.y; o[2] = (__bf16)a.z; o[3] = (__bf16)a.w;
  o[4] = (__bf16)b.x; o[5] = (__bf16)b.y; o[6] = (__bf16)b.z; o[7] = (__bf16)b.w;
  *(v8bf*)(Y + i) = o;
}

// ---------------- squared norms: one wave per row ----------------
__global__ void sqnorm_kernel(const float* __restrict__ X, float* __restrict__ xx) {
  int wave = threadIdx.x >> 5, lane = threadIdx.x & 31;
  int row = blockIdx.x * 8 + wave;
  const float* p = X + (size_t)row * Dn;
  float s = 0.0f;
  for (int k = lane; k < Dn; k += 32) { float v = p[k]; s += v * v; }
  for (int m = 16; m; m >>= 1) s += __shfl_xor(s, m, 32);
  if (lane == 0) xx[row] = s;
}

// ---------------- WMMA frag loaders ----------------
// 16-bit A 16x32 layout (ISA 7.12.2): lane&15 = M-row, lane>>4 selects K-half.
// Per lane: K = kb..kb+7 and kb+16..kb+23, kb = kc*32 + (lane>>4)*8.
// B (KxN) mirrors it (lane&15 = N-col, same packed K), so one loader serves both
// operands of X*X^T.
__device__ __forceinline__ v16bf load_frag_f32(const float* __restrict__ X, int row,
                                               int kc, int khalf) {
  const float* p = X + (size_t)row * Dn + kc * 32 + khalf * 8;
  float4 f0 = *(const float4*)(p);
  float4 f1 = *(const float4*)(p + 4);
  float4 f2 = *(const float4*)(p + 16);
  float4 f3 = *(const float4*)(p + 20);
  v16bf f;
  f[0]  = (__bf16)f0.x; f[1]  = (__bf16)f0.y; f[2]  = (__bf16)f0.z; f[3]  = (__bf16)f0.w;
  f[4]  = (__bf16)f1.x; f[5]  = (__bf16)f1.y; f[6]  = (__bf16)f1.z; f[7]  = (__bf16)f1.w;
  f[8]  = (__bf16)f2.x; f[9]  = (__bf16)f2.y; f[10] = (__bf16)f2.z; f[11] = (__bf16)f2.w;
  f[12] = (__bf16)f3.x; f[13] = (__bf16)f3.y; f[14] = (__bf16)f3.z; f[15] = (__bf16)f3.w;
  return f;
}

__device__ __forceinline__ v16bf load_frag_bf(const __bf16* __restrict__ X, int row,
                                              int kc, int khalf) {
  const __bf16* p = X + (size_t)row * Dn + kc * 32 + khalf * 8;
  v8bf lo = *(const v8bf*)(p);        // K = kb .. kb+7   (16 B)
  v8bf hi = *(const v8bf*)(p + 16);   // K = kb+16..kb+23 (16 B)
  return __builtin_shufflevector(lo, hi, 0, 1, 2, 3, 4, 5, 6, 7,
                                 8, 9, 10, 11, 12, 13, 14, 15);
}

template <bool STAGED>
__device__ __forceinline__ v16bf load_frag(const float* __restrict__ Xf,
                                           const __bf16* __restrict__ Xb, int row,
                                           int kc, int khalf) {
  if (STAGED) return load_frag_bf(Xb, row, kc, khalf);
  return load_frag_f32(Xf, row, kc, khalf);
}

// ---------------- pass 1: D tiles -> deterministic row sums ----------------
// grid (64,8): blockIdx.x = 128-row block (8 waves x 16 rows), blockIdx.y = j-split
template <bool STAGED>
__global__ void __launch_bounds__(256)
dist_rowsum_kernel(const float* __restrict__ Xf, const __bf16* __restrict__ Xb,
                   const float* __restrict__ xx, float* __restrict__ rowPart) {
  int wave = threadIdx.x >> 5, lane = threadIdx.x & 31;
  int khalf = lane >> 4, n = lane & 15;
  int r0 = blockIdx.x * 128 + wave * 16;
  int jbase = blockIdx.y * 1024;

  v16bf afr[8];
#pragma unroll
  for (int kc = 0; kc < 8; kc++) afr[kc] = load_frag<STAGED>(Xf, Xb, r0 + n, kc, khalf);
  float xi[8];
#pragma unroll
  for (int r = 0; r < 8; r++) xi[r] = xx[r0 + 8 * khalf + r];

  float rp[8];
#pragma unroll
  for (int r = 0; r < 8; r++) rp[r] = 0.0f;

  for (int jt = 0; jt < 64; jt++) {
    int j0 = jbase + jt * 16;
    float xj = xx[j0 + n];
    // batch all B-frags first: 16 loads in flight, then the 8-WMMA chain drains
    v16bf bfr[8];
#pragma unroll
    for (int kc = 0; kc < 8; kc++) bfr[kc] = load_frag<STAGED>(Xf, Xb, j0 + n, kc, khalf);
    v8f acc = {};
#pragma unroll
    for (int kc = 0; kc < 8; kc++)
      acc = __builtin_amdgcn_wmma_f32_16x16x32_bf16(false, afr[kc], false, bfr[kc],
                                                    (short)0, acc, false, false);
#pragma unroll
    for (int r = 0; r < 8; r++) {
      float dsq = xi[r] + xj - 2.0f * acc[r];
      rp[r] += sqrtf(fmaxf(dsq, 1e-8f));
    }
  }
  // reduce over the 16 columns (lanes within each half-wave share the same rows)
#pragma unroll
  for (int r = 0; r < 8; r++) {
    float v = rp[r];
    v += __shfl_xor(v, 1, 32); v += __shfl_xor(v, 2, 32);
    v += __shfl_xor(v, 4, 32); v += __shfl_xor(v, 8, 32);
    if (n == 0) rowPart[(size_t)blockIdx.y * Bn + r0 + 8 * khalf + r] = v;
  }
}

// ---------------- fold j-split partials into row sums ----------------
__global__ void reduce_rows_kernel(float* __restrict__ ws) {
  int i = blockIdx.x * 256 + threadIdx.x;
  float a = 0.0f, b = 0.0f;
#pragma unroll
  for (int s = 0; s < 8; s++) {
    a += ws[OFF_ROWAP + s * Bn + i];
    b += ws[OFF_ROWBP + s * Bn + i];
  }
  ws[OFF_ROWA + i] = a;
  ws[OFF_ROWB + i] = b;
}

// ---------------- global sums of row sums (1 block) ----------------
__global__ void gsum_kernel(float* __restrict__ ws) {
  __shared__ float sa[256], sb[256];
  int tid = threadIdx.x;
  float a = 0.0f, b = 0.0f;
  for (int i = tid; i < Bn; i += 256) { a += ws[OFF_ROWA + i]; b += ws[OFF_ROWB + i]; }
  sa[tid] = a; sb[tid] = b;
  __syncthreads();
  for (int s = 128; s; s >>= 1) {
    if (tid < s) { sa[tid] += sa[tid + s]; sb[tid] += sb[tid + s]; }
    __syncthreads();
  }
  if (tid == 0) { ws[OFF_SCAL + 0] = sa[0]; ws[OFF_SCAL + 1] = sb[0]; }
}

// ---------------- pass 2: recompute D_A,D_B, center, accumulate moments ----------------
template <bool STAGED>
__global__ void __launch_bounds__(256)
dcor_accum_kernel(const float* __restrict__ XAf, const float* __restrict__ XBf,
                  const __bf16* __restrict__ XAb, const __bf16* __restrict__ XBb,
                  float* __restrict__ ws) {
  const float* xxA  = ws + OFF_XXA;
  const float* xxB  = ws + OFF_XXB;
  const float* rowA = ws + OFF_ROWA;
  const float* rowB = ws + OFF_ROWB;
  const float invN  = 1.0f / (float)Bn;
  const float invN2 = invN * invN;
  const float ga = ws[OFF_SCAL + 0] * invN2;
  const float gb = ws[OFF_SCAL + 1] * invN2;

  int wave = threadIdx.x >> 5, lane = threadIdx.x & 31;
  int khalf = lane >> 4, n = lane & 15;
  int r0 = blockIdx.x * 128 + wave * 16;
  int jbase = blockIdx.y * 1024;

  v16bf afA[8], afB[8];
#pragma unroll
  for (int kc = 0; kc < 8; kc++) {
    afA[kc] = load_frag<STAGED>(XAf, XAb, r0 + n, kc, khalf);
    afB[kc] = load_frag<STAGED>(XBf, XBb, r0 + n, kc, khalf);
  }
  float xiA[8], xiB[8], miA[8], miB[8];
#pragma unroll
  for (int r = 0; r < 8; r++) {
    int i = r0 + 8 * khalf + r;
    xiA[r] = xxA[i]; xiB[r] = xxB[i];
    miA[r] = rowA[i] * invN; miB[r] = rowB[i] * invN;
  }

  float s0 = 0.0f, s1 = 0.0f, s2 = 0.0f;
  for (int jt = 0; jt < 64; jt++) {
    int j0 = jbase + jt * 16;
    float xjA = xxA[j0 + n], xjB = xxB[j0 + n];
    float mjA = rowA[j0 + n] * invN, mjB = rowB[j0 + n] * invN;
    v8f aA = {}, aB = {};
    {
      v16bf bfr[8];
#pragma unroll
      for (int kc = 0; kc < 8; kc++) bfr[kc] = load_frag<STAGED>(XAf, XAb, j0 + n, kc, khalf);
#pragma unroll
      for (int kc = 0; kc < 8; kc++)
        aA = __builtin_amdgcn_wmma_f32_16x16x32_bf16(false, afA[kc], false, bfr[kc],
                                                     (short)0, aA, false, false);
    }
    {
      v16bf bfr[8];
#pragma unroll
      for (int kc = 0; kc < 8; kc++) bfr[kc] = load_frag<STAGED>(XBf, XBb, j0 + n, kc, khalf);
#pragma unroll
      for (int kc = 0; kc < 8; kc++)
        aB = __builtin_amdgcn_wmma_f32_16x16x32_bf16(false, afB[kc], false, bfr[kc],
                                                     (short)0, aB, false, false);
    }
#pragma unroll
    for (int r = 0; r < 8; r++) {
      float da = sqrtf(fmaxf(xiA[r] + xjA - 2.0f * aA[r], 1e-8f));
      float db = sqrtf(fmaxf(xiB[r] + xjB - 2.0f * aB[r], 1e-8f));
      float am = da - miA[r] - mjA + ga;
      float bm = db - miB[r] - mjB + gb;
      s0 += am * bm; s1 += am * am; s2 += bm * bm;
    }
  }
  for (int m = 16; m; m >>= 1) {
    s0 += __shfl_xor(s0, m, 32);
    s1 += __shfl_xor(s1, m, 32);
    s2 += __shfl_xor(s2, m, 32);
  }
  if (lane == 0) {
    int gid = ((int)blockIdx.y * 64 + (int)blockIdx.x) * 8 + wave;
    ws[OFF_WPART + 3 * gid + 0] = s0;
    ws[OFF_WPART + 3 * gid + 1] = s1;
    ws[OFF_WPART + 3 * gid + 2] = s2;
  }
}

// ---------------- Cox: max(risk) for both endpoints (1 block) ----------------
__global__ void riskmax_kernel(const float* __restrict__ r_os,
                               const float* __restrict__ r_rfs,
                               float* __restrict__ ws) {
  __shared__ float s1[256], s2[256];
  int tid = threadIdx.x;
  float m1 = -INFINITY, m2 = -INFINITY;
  for (int i = tid; i < Bn; i += 256) {
    m1 = fmaxf(m1, r_os[i]);
    m2 = fmaxf(m2, r_rfs[i]);
  }
  s1[tid] = m1; s2[tid] = m2;
  __syncthreads();
  for (int s = 128; s; s >>= 1) {
    if (tid < s) { s1[tid] = fmaxf(s1[tid], s1[tid + s]); s2[tid] = fmaxf(s2[tid], s2[tid + s]); }
    __syncthreads();
  }
  if (tid == 0) { ws[OFF_SCAL + 2] = s1[0]; ws[OFF_SCAL + 3] = s2[0]; }
}

// ---------------- Cox: per-time-bin histogram (ties == equal integer times) ----------------
__global__ void cox_hist_kernel(const float* __restrict__ risk,
                                const int* __restrict__ evt,
                                const int* __restrict__ tm,
                                float* __restrict__ ws, int ep, int maxIdx) {
  int i = blockIdx.x * 256 + threadIdx.x;
  float maxr = ws[OFF_SCAL + maxIdx];
  float* h = ws + OFF_HIST + ep * 4096;
  float r = risk[i];
  int t = tm[i];
  float er = expf(r - maxr);
  atomicAdd(&h[t], er);                         // S[t]  : risk-set mass at time t
  if (evt[i]) {
    atomicAdd(&h[1024 + t], er);                // E[t]  : tied event exp-risk (tes)
    atomicAdd(&h[2048 + t], 1.0f);              // d[t]  : tie count
    atomicAdd(&h[3072 + t], r);                 // R[t]  : sum of event raw risks
  }
}

// ---------------- final: reduce dcor moments, finish both Cox losses, combine ----------------
__global__ void __launch_bounds__(1024)
final_kernel(float* __restrict__ ws, const float* __restrict__ intra_wsi,
             const float* __restrict__ intra_ct, float* __restrict__ out) {
  __shared__ float red0[1024], red1[1024], red2[1024];
  __shared__ float rss[NBINS];
  __shared__ float sres[8];
  int tid = threadIdx.x;

  float a = 0.0f, b = 0.0f, c = 0.0f;
  for (int k = tid; k < 4096; k += 1024) {
    a += ws[OFF_WPART + 3 * k + 0];
    b += ws[OFF_WPART + 3 * k + 1];
    c += ws[OFF_WPART + 3 * k + 2];
  }
  red0[tid] = a; red1[tid] = b; red2[tid] = c;
  __syncthreads();
  for (int s = 512; s; s >>= 1) {
    if (tid < s) { red0[tid] += red0[tid + s]; red1[tid] += red1[tid + s]; red2[tid] += red2[tid + s]; }
    __syncthreads();
  }
  if (tid == 0) { sres[0] = red0[0]; sres[1] = red1[0]; sres[2] = red2[0]; }
  __syncthreads();

  for (int ep = 0; ep < 2; ep++) {
    const float* h = ws + OFF_HIST + ep * 4096;
    float maxr = ws[OFF_SCAL + 2 + ep];
    if (tid == 0) {                      // suffix sum over descending time
      float run = 0.0f;
      for (int t = NBINS - 1; t >= 0; --t) { run += h[t]; rss[t] = run; }
    }
    __syncthreads();
    float cont = 0.0f, nev = 0.0f;
    if (tid < NBINS) {
      float R = h[3072 + tid], D = h[2048 + tid], E = h[1024 + tid];
      int di = (int)(D + 0.5f);
      if (di > 0) {
        float s = 0.0f;
        for (int l = 0; l < di; l++)
          s += logf(rss[tid] - ((float)l / D) * E + 1e-12f);
        cont = R - D * maxr - s;
        nev = D;
      }
    }
    red0[tid] = cont; red1[tid] = nev;
    __syncthreads();
    for (int s = 512; s; s >>= 1) {
      if (tid < s) { red0[tid] += red0[tid + s]; red1[tid] += red1[tid + s]; }
      __syncthreads();
    }
    if (tid == 0) { sres[3 + 2 * ep] = red0[0]; sres[4 + 2 * ep] = red1[0]; }
    __syncthreads();
  }

  if (tid == 0) {
    float N2 = (float)Bn * (float)Bn;
    float dcov = sres[0] / N2, va = sres[1] / N2, vb = sres[2] / N2;
    float l_cca = 1.0f - dcov / sqrtf(fmaxf(va * vb, 1e-8f));
    float l_os  = -sres[3] / (sres[4] + 1e-12f);
    float l_rfs = -sres[5] / (sres[6] + 1e-12f);
    float l_intra = intra_wsi[0] + intra_ct[0];
    out[0] = 1.0f * l_os + 1.0f * l_rfs + 1.0f * (0.1f * l_cca + 0.05f * l_intra);
  }
}

extern "C" void kernel_launch(void* const* d_in, const int* in_sizes, int n_in,
                              void* d_out, int out_size, void* d_ws, size_t ws_size,
                              hipStream_t stream) {
  const float* risk_os   = (const float*)d_in[0];
  const float* risk_rfs  = (const float*)d_in[1];
  const float* z_ct      = (const float*)d_in[2];
  const float* z_wsi     = (const float*)d_in[3];
  const float* intra_wsi = (const float*)d_in[4];
  const float* intra_ct  = (const float*)d_in[5];
  const int*   evt_os    = (const int*)d_in[6];
  const int*   tm_os     = (const int*)d_in[7];
  const int*   evt_rfs   = (const int*)d_in[8];
  const int*   tm_rfs    = (const int*)d_in[9];
  float* ws  = (float*)d_ws;
  float* out = (float*)d_out;

  const bool staged = (ws_size >= STAGED_BYTES_NEEDED);
  __bf16* bfA = (__bf16*)((char*)d_ws + OFF_BF16A_BYTES);
  __bf16* bfB = (__bf16*)((char*)d_ws + OFF_BF16B_BYTES);

  zero_kernel<<<256, 256, 0, stream>>>(ws);
  sqnorm_kernel<<<1024, 256, 0, stream>>>(z_ct,  ws + OFF_XXA);
  sqnorm_kernel<<<1024, 256, 0, stream>>>(z_wsi, ws + OFF_XXB);

  if (staged) {
    cvt_bf16_kernel<<<1024, 256, 0, stream>>>(z_ct,  bfA);
    cvt_bf16_kernel<<<1024, 256, 0, stream>>>(z_wsi, bfB);
    dist_rowsum_kernel<true><<<dim3(64, 8), 256, 0, stream>>>(z_ct,  bfA, ws + OFF_XXA, ws + OFF_ROWAP);
    dist_rowsum_kernel<true><<<dim3(64, 8), 256, 0, stream>>>(z_wsi, bfB, ws + OFF_XXB, ws + OFF_ROWBP);
  } else {
    dist_rowsum_kernel<false><<<dim3(64, 8), 256, 0, stream>>>(z_ct,  nullptr, ws + OFF_XXA, ws + OFF_ROWAP);
    dist_rowsum_kernel<false><<<dim3(64, 8), 256, 0, stream>>>(z_wsi, nullptr, ws + OFF_XXB, ws + OFF_ROWBP);
  }

  reduce_rows_kernel<<<32, 256, 0, stream>>>(ws);
  gsum_kernel<<<1, 256, 0, stream>>>(ws);

  if (staged) {
    dcor_accum_kernel<true><<<dim3(64, 8), 256, 0, stream>>>(z_ct, z_wsi, bfA, bfB, ws);
  } else {
    dcor_accum_kernel<false><<<dim3(64, 8), 256, 0, stream>>>(z_ct, z_wsi, nullptr, nullptr, ws);
  }

  riskmax_kernel<<<1, 256, 0, stream>>>(risk_os, risk_rfs, ws);
  cox_hist_kernel<<<32, 256, 0, stream>>>(risk_os,  evt_os,  tm_os,  ws, 0, 2);
  cox_hist_kernel<<<32, 256, 0, stream>>>(risk_rfs, evt_rfs, tm_rfs, ws, 1, 3);
  final_kernel<<<1, 1024, 0, stream>>>(ws, intra_wsi, intra_ct, out);
}
